// Attention_39943195852901
// MI455X (gfx1250) — matrix operand order repaired and checked by reading, not compiled, not run on hardware
//
#include <hip/hip_runtime.h>
#include <hip/hip_fp16.h>

typedef __attribute__((ext_vector_type(16))) _Float16 v16h;
typedef __attribute__((ext_vector_type(8)))  _Float16 v8h;
typedef __attribute__((ext_vector_type(8)))  float    v8f;

#define WMMA_F32_F16(a, b, c) \
  __builtin_amdgcn_wmma_f32_16x16x32_f16(false, (a), false, (b), (short)0, (c), false, false)

constexpr int Bsz  = 4;
constexpr int Nseq = 2048;
constexpr int Cdim = 1024;
constexpr int Hh   = 16;
constexpr int Dh   = 64;

// Build a 16-half fragment from two contiguous 16-byte chunks
// (each chunk -> ds_load_b128 / global_load_b128).
__device__ __forceinline__ v16h ldfrag2(const _Float16* p0, const _Float16* p1) {
  v8h a = *(const v8h*)p0;
  v8h b = *(const v8h*)p1;
  return __builtin_shufflevector(a, b, 0, 1, 2, 3, 4, 5, 6, 7,
                                 8, 9, 10, 11, 12, 13, 14, 15);
}

// ---------------------------------------------------------------------------
// Kernel 1: QKV projection GEMM, fp32 in -> f16 out.
//   mode 0 (Q): Qh[b][h][n][d]   (pre-scaled by 1/sqrt(D) = 0.125)
//   mode 1 (K): Kh[b][h][n][d]   (natural: d is WMMA-K for QK^T)
//   mode 2 (V): Vt[b][h][d][n]   (transposed: key is WMMA-K for PV)
// Block tile 256x128, 8 waves (4x2) of 64x64 each -> 16 WMMAs per wave per
// k-step against 16 ds_load_b128 (1.0 load/WMMA). W tile stored TRANSPOSED
// in LDS (k contiguous) so B-fragments are two ds_load_b128.
// ---------------------------------------------------------------------------
__global__ __launch_bounds__(256) void qkv_gemm(
    const float* __restrict__ x,
    const float* __restrict__ Wq, const float* __restrict__ Wk,
    const float* __restrict__ Wv,
    _Float16* __restrict__ Qh, _Float16* __restrict__ Kh,
    _Float16* __restrict__ Vt) {
  constexpr int TK = 32;
  constexpr int XS = 40;  // A-tile LDS row stride (halfs), row-major [row][k]
  constexpr int WT = 40;  // W-tile LDS row stride (halfs), TRANSPOSED [n][k]
  __shared__ _Float16 Xs[256 * XS];
  __shared__ _Float16 Wt[128 * WT];

  const int t    = threadIdx.x;
  const int lane = t & 31;
  const int wave = t >> 5;
  const int half = lane >> 4;
  const int m    = lane & 15;
  const int wm   = wave >> 1;  // 0..3 (M)
  const int wn   = wave & 1;   // 0..1 (N)
  const int mode = blockIdx.z;
  const float* W = (mode == 0) ? Wq : (mode == 1) ? Wk : Wv;
  const int m0 = blockIdx.y * 256;
  const int n0 = blockIdx.x * 128;

  v8f acc[4][4] = {};

  for (int k0 = 0; k0 < Cdim; k0 += TK) {
    // A tile: 256 rows x 32 k, float4 global loads, contiguous f16 LDS rows
#pragma unroll
    for (int p = 0; p < 8; ++p) {
      int row = p * 32 + (t >> 3);
      int col = (t & 7) * 4;
      float4 f4 = *(const float4*)(x + (size_t)(m0 + row) * Cdim + k0 + col);
      _Float16* dst = &Xs[row * XS + col];
      dst[0] = (_Float16)f4.x; dst[1] = (_Float16)f4.y;
      dst[2] = (_Float16)f4.z; dst[3] = (_Float16)f4.w;
    }
    // W tile: 32 k x 128 n, float4 global loads (along n), transposed to LDS
#pragma unroll
    for (int p = 0; p < 4; ++p) {
      int krow = p * 8 + (t >> 5);
      int col  = (t & 31) * 4;
      float4 f4 = *(const float4*)(W + (size_t)(k0 + krow) * Cdim + n0 + col);
      Wt[(col + 0) * WT + krow] = (_Float16)f4.x;
      Wt[(col + 1) * WT + krow] = (_Float16)f4.y;
      Wt[(col + 2) * WT + krow] = (_Float16)f4.z;
      Wt[(col + 3) * WT + krow] = (_Float16)f4.w;
    }
    __syncthreads();

    v16h af[4];
#pragma unroll
    for (int ms = 0; ms < 4; ++ms) {
      const _Float16* pr = &Xs[(wm * 64 + ms * 16 + m) * XS + half * 16];
      af[ms] = ldfrag2(pr, pr + 8);
    }
#pragma unroll
    for (int ns = 0; ns < 4; ++ns) {
      const _Float16* pc = &Wt[(wn * 64 + ns * 16 + m) * WT + half * 16];
      v16h bf = ldfrag2(pc, pc + 8);
#pragma unroll
      for (int ms = 0; ms < 4; ++ms)
        acc[ms][ns] = WMMA_F32_F16(af[ms], bf, acc[ms][ns]);
    }
    __syncthreads();
  }

  // Epilogue: C/D layout (VGPR r -> row half*8+r, lane%16 -> col)
  _Float16* dstQK = (mode == 0) ? Qh : Kh;
  const float scale = (mode == 0) ? 0.125f : 1.0f;
#pragma unroll
  for (int ms = 0; ms < 4; ++ms) {
    int rbase = m0 + wm * 64 + ms * 16 + half * 8;  // multiple of 8
    int b = rbase >> 11, n = rbase & (Nseq - 1);
#pragma unroll
    for (int ns = 0; ns < 4; ++ns) {
      int cg = n0 + wn * 64 + ns * 16 + m;
      int h = cg >> 6, d = cg & (Dh - 1);
      if (mode == 2) {
        // 8 consecutive n per lane -> one 16B store
        v8h pk;
#pragma unroll
        for (int r = 0; r < 8; ++r) pk[r] = (_Float16)acc[ms][ns][r];
        *(v8h*)&Vt[(((size_t)(b * Hh + h)) * Dh + d) * Nseq + n] = pk;
      } else {
        size_t base = (((size_t)(b * Hh + h)) * Nseq + n) * Dh + d;
#pragma unroll
        for (int r = 0; r < 8; ++r)
          dstQK[base + (size_t)r * Dh] = (_Float16)(acc[ms][ns][r] * scale);
      }
    }
  }
}

// ---------------------------------------------------------------------------
// Kernel 2: flash attention. Block = 128 query rows of one (b,h); 8 waves,
// each owning a 16-row strip. 64-key K (natural) / V^T tiles in LDS shared by
// all waves; fp32 online softmax with wave32 half-group shuffles; P bounced
// through per-wave LDS strip to A-fragment layout. All fragments b128 loads.
// ---------------------------------------------------------------------------
__global__ __launch_bounds__(256) void flash_attn(
    const _Float16* __restrict__ Qh, const _Float16* __restrict__ Kh,
    const _Float16* __restrict__ Vt, _Float16* __restrict__ Oh) {
  constexpr int ST = 72;  // LDS row stride in halfs (64 + 8 pad)
  __shared__ _Float16 Kls[64 * ST];      // K tile:   [key][d]
  __shared__ _Float16 Vtl[64 * ST];      // V^T tile: [d][key]
  __shared__ _Float16 Ps[8 * 16 * ST];   // per-wave P strips [row][key]

  const int t    = threadIdx.x;
  const int lane = t & 31;
  const int w    = t >> 5;
  const int half = lane >> 4;
  const int m    = lane & 15;
  const int b    = blockIdx.z;
  const int h    = blockIdx.y;
  const int bh   = b * Hh + h;
  const int q0   = blockIdx.x * 128;
  const int qrow = q0 + w * 16 + m;

  // Q fragments: two 16x32 chunks over D=64, contiguous global b128 loads
  v16h aq[2];
  const _Float16* qg = Qh + ((size_t)bh * Nseq + qrow) * Dh;
#pragma unroll
  for (int c = 0; c < 2; ++c)
    aq[c] = ldfrag2(qg + c * 32 + half * 16, qg + c * 32 + 8 + half * 16);

  v8f vo[4] = {};
  float mrow[8], lrow[8];
#pragma unroll
  for (int r = 0; r < 8; ++r) { mrow[r] = -3.0e38f; lrow[r] = 0.0f; }

  unsigned int* Kls_u = (unsigned int*)Kls;
  unsigned int* Vtl_u = (unsigned int*)Vtl;
  const unsigned int* kg_u = (const unsigned int*)(Kh + (size_t)bh * Nseq * Dh);
  const unsigned int* vg_u = (const unsigned int*)(Vt + (size_t)bh * Dh * Nseq);

  const int crow = t >> 2;       // 0..63
  const int ccol = (t & 3) * 8;  // uint offset within row

  for (int kt = 0; kt < Nseq; kt += 64) {
    // K tile: 64 keys x 64 d (rows contiguous in global)
#pragma unroll
    for (int i = 0; i < 8; ++i)
      Kls_u[crow * (ST / 2) + ccol + i] =
          kg_u[(size_t)(kt + crow) * (Dh / 2) + ccol + i];
    // V^T tile: 64 d x 64 keys (rows contiguous in global)
#pragma unroll
    for (int i = 0; i < 8; ++i)
      Vtl_u[crow * (ST / 2) + ccol + i] =
          vg_u[(size_t)crow * (Nseq / 2) + (kt >> 1) + ccol + i];
    if (kt + 64 < Nseq) {  // prefetch next tiles (global_prefetch_b8)
      __builtin_prefetch(
          (const void*)(kg_u + (size_t)(kt + 64 + crow) * (Dh / 2) + ccol), 0, 1);
      __builtin_prefetch(
          (const void*)(vg_u + (size_t)crow * (Nseq / 2) + ((kt + 64) >> 1) + ccol),
          0, 1);
    }
    __syncthreads();

    // S = Q K^T (16 x 64 per wave; WMMA-K = d); B-frag = Kls[key][d-chunk]
    v8f s[4] = {};
#pragma unroll
    for (int c = 0; c < 2; ++c)
#pragma unroll
      for (int nt = 0; nt < 4; ++nt) {
        const _Float16* pk = &Kls[(nt * 16 + m) * ST + c * 32 + half * 16];
        v16h bk = ldfrag2(pk, pk + 8);
        s[nt] = WMMA_F32_F16(aq[c], bk, s[nt]);
      }

    // online softmax per owned row (rows half*8+r; 16-lane reductions)
#pragma unroll
    for (int r = 0; r < 8; ++r) {
      float tm = -3.0e38f;
#pragma unroll
      for (int nt = 0; nt < 4; ++nt) tm = fmaxf(tm, s[nt][r]);
#pragma unroll
      for (int mk = 1; mk <= 8; mk <<= 1)
        tm = fmaxf(tm, __shfl_xor(tm, mk, 32));
      float mnew = fmaxf(mrow[r], tm);
      float sc   = __expf(mrow[r] - mnew);
      float rs   = 0.0f;
#pragma unroll
      for (int nt = 0; nt < 4; ++nt) {
        float p = __expf(s[nt][r] - mnew);
        s[nt][r] = p;
        rs += p;
      }
#pragma unroll
      for (int mk = 1; mk <= 8; mk <<= 1) rs += __shfl_xor(rs, mk, 32);
      lrow[r] = lrow[r] * sc + rs;
      mrow[r] = mnew;
#pragma unroll
      for (int dt = 0; dt < 4; ++dt) vo[dt][r] *= sc;
    }

    // P: C/D layout -> per-wave LDS strip (same wave; compiler waits dscnt)
    _Float16* Pw = &Ps[w * 16 * ST];
#pragma unroll
    for (int nt = 0; nt < 4; ++nt)
#pragma unroll
      for (int r = 0; r < 8; ++r)
        Pw[(half * 8 + r) * ST + nt * 16 + m] = (_Float16)s[nt][r];

    // reload P as A-fragments (contiguous b128 chunks)
    v16h ap[2];
#pragma unroll
    for (int c = 0; c < 2; ++c) {
      const _Float16* pp = &Pw[m * ST + c * 32 + half * 16];
      ap[c] = ldfrag2(pp, pp + 8);
    }

    // O += P V (WMMA-K = key); B-frag = Vtl[d-col][key-chunk]
#pragma unroll
    for (int dt = 0; dt < 4; ++dt)
#pragma unroll
      for (int c = 0; c < 2; ++c) {
        const _Float16* pv = &Vtl[(dt * 16 + m) * ST + c * 32 + half * 16];
        v16h bv = ldfrag2(pv, pv + 8);
        vo[dt] = WMMA_F32_F16(ap[c], bv, vo[dt]);
      }
    __syncthreads();
  }

  // normalize and store merged-head O as f16 [B][N][C]
#pragma unroll
  for (int dt = 0; dt < 4; ++dt)
#pragma unroll
    for (int r = 0; r < 8; ++r) {
      float inv = 1.0f / lrow[r];
      int rowg = q0 + w * 16 + half * 8 + r;
      int colg = h * Dh + dt * 16 + m;
      Oh[((size_t)b * Nseq + rowg) * Cdim + colg] =
          (_Float16)(vo[dt][r] * inv);
    }
}

// ---------------------------------------------------------------------------
// Kernel 3: out = O @ Wo + bo, f16 A / f16-converted transposed-W, fp32 out.
// Same 256x128 block / 64x64 wave tiling as kernel 1.
// ---------------------------------------------------------------------------
__global__ __launch_bounds__(256) void out_proj(
    const _Float16* __restrict__ Oh, const float* __restrict__ Wo,
    const float* __restrict__ bo, float* __restrict__ out) {
  constexpr int TK = 32;
  constexpr int XS = 40;
  constexpr int WT = 40;
  __shared__ _Float16 Xs[256 * XS];
  __shared__ _Float16 Wt[128 * WT];

  const int t    = threadIdx.x;
  const int lane = t & 31;
  const int wave = t >> 5;
  const int half = lane >> 4;
  const int m    = lane & 15;
  const int wm   = wave >> 1;
  const int wn   = wave & 1;
  const int m0 = blockIdx.y * 256;
  const int n0 = blockIdx.x * 128;

  v8f acc[4][4] = {};

  unsigned int* Xs_u = (unsigned int*)Xs;
  const unsigned int* og_u = (const unsigned int*)Oh;

  for (int k0 = 0; k0 < Cdim; k0 += TK) {
    // A tile (already f16): 256 rows x 32 halfs = 16 uints per row
#pragma unroll
    for (int p = 0; p < 2; ++p) {
      int row = p * 128 + (t >> 1);
      int ug  = (t & 1) * 8;
#pragma unroll
      for (int i = 0; i < 8; ++i)
        Xs_u[row * (XS / 2) + ug + i] =
            og_u[(size_t)(m0 + row) * (Cdim / 2) + (k0 >> 1) + ug + i];
    }
    // W tile: 32 k x 128 n, transposed into LDS [n][k]
#pragma unroll
    for (int p = 0; p < 4; ++p) {
      int krow = p * 8 + (t >> 5);
      int col  = (t & 31) * 4;
      float4 f4 = *(const float4*)(Wo + (size_t)(k0 + krow) * Cdim + n0 + col);
      Wt[(col + 0) * WT + krow] = (_Float16)f4.x;
      Wt[(col + 1) * WT + krow] = (_Float16)f4.y;
      Wt[(col + 2) * WT + krow] = (_Float16)f4.z;
      Wt[(col + 3) * WT + krow] = (_Float16)f4.w;
    }
    __syncthreads();

    v16h af[4];
#pragma unroll
    for (int ms = 0; ms < 4; ++ms) {
      const _Float16* pr = &Xs[(wm * 64 + ms * 16 + m) * XS + half * 16];
      af[ms] = ldfrag2(pr, pr + 8);
    }
#pragma unroll
    for (int ns = 0; ns < 4; ++ns) {
      const _Float16* pc = &Wt[(wn * 64 + ns * 16 + m) * WT + half * 16];
      v16h bf = ldfrag2(pc, pc + 8);
#pragma unroll
      for (int ms = 0; ms < 4; ++ms)
        acc[ms][ns] = WMMA_F32_F16(af[ms], bf, acc[ms][ns]);
    }
    __syncthreads();
  }

#pragma unroll
  for (int ms = 0; ms < 4; ++ms) {
    int rbase = m0 + wm * 64 + ms * 16 + half * 8;
#pragma unroll
    for (int ns = 0; ns < 4; ++ns) {
      int cg = n0 + wn * 64 + ns * 16 + m;
      float bias = bo[cg];
      size_t base = (size_t)rbase * Cdim + cg;
#pragma unroll
      for (int r = 0; r < 8; ++r)
        out[base + (size_t)r * Cdim] = acc[ms][ns][r] + bias;
    }
  }
}

// ---------------------------------------------------------------------------
extern "C" void kernel_launch(void* const* d_in, const int* in_sizes, int n_in,
                              void* d_out, int out_size, void* d_ws,
                              size_t ws_size, hipStream_t stream) {
  (void)in_sizes; (void)n_in; (void)out_size; (void)ws_size;
  const float* x  = (const float*)d_in[0];
  const float* Wq = (const float*)d_in[1];
  const float* Wk = (const float*)d_in[2];
  const float* Wv = (const float*)d_in[3];
  const float* Wo = (const float*)d_in[4];
  const float* bo = (const float*)d_in[5];
  float* out = (float*)d_out;

  const size_t elems = (size_t)Bsz * Hh * Nseq * Dh;  // 8,388,608
  _Float16* Qh = (_Float16*)d_ws;
  _Float16* Kh = Qh + elems;
  _Float16* Vt = Kh + elems;
  _Float16* Oh = Vt + elems;  // total ws use: 4 * 16 MB = 64 MB

  dim3 blk(256);
  qkv_gemm<<<dim3(Cdim / 128, (Bsz * Nseq) / 256, 3), blk, 0, stream>>>(
      x, Wq, Wk, Wv, Qh, Kh, Vt);
  flash_attn<<<dim3(Nseq / 128, Hh, Bsz), blk, 0, stream>>>(Qh, Kh, Vt, Oh);
  out_proj<<<dim3(Cdim / 128, (Bsz * Nseq) / 256, 1), blk, 0, stream>>>(
      Oh, Wo, bo, out);
}
